// ResnetGru_85512798863834
// MI455X (gfx1250) — compile-verified, matrix operand order checked by reading, not run on hardware
//
#include <hip/hip_runtime.h>
#include <math.h>

// ---------------------------------------------------------------------------
// MI455X (gfx1250) implementation of ResNet18 + triangular GRU decode.
//  * All GEMM-shaped work uses v_wmma_f32_16x16x32_f16 (wave32 WMMA).
//  * Fragment layout fact: element pairs v=0..3 of a lane cover 8 consecutive
//    halves at kc+half8, v=4..7 the 8 halves at kc+16+half8 -> a whole f16
//    fragment is two aligned 16B chunks (ds_load_b128 / float4+cvt_pk).
//  * GRU: W_hh sliced into LDS across 32 WGs (48KB/WG), h staged to LDS as f16
//    per cell, one grid barrier/cell, double-buffered h, 20100 sequential cells.
// ---------------------------------------------------------------------------

typedef __attribute__((ext_vector_type(16))) _Float16 v16h;
typedef __attribute__((ext_vector_type(8)))  float    v8f;
typedef __attribute__((ext_vector_type(8)))  unsigned v8u;
typedef __attribute__((ext_vector_type(4)))  unsigned v4u;
typedef __attribute__((ext_vector_type(4)))  float    v4f;

#define TMAX    200
#define HDIM    512
#define NB      4
#define GRU_WGS 32

// ISA §7.12.2: element pair v (v=0..7) of lane L holds K = {k0, k0+1},
// k0 = 16*(v/4) + 8*(L/16) + 2*(v%4).  Row/col = L%16.
__device__ __forceinline__ int kbase(int v, int half8) {
  return ((v >> 2) << 4) + half8 + ((v & 3) << 1);
}

__device__ __forceinline__ unsigned pk2(float lo, float hi) {
  unsigned short a = __builtin_bit_cast(unsigned short, (_Float16)lo);
  unsigned short b = __builtin_bit_cast(unsigned short, (_Float16)hi);
  return (unsigned)a | ((unsigned)b << 16);
}

// pack 8 consecutive f32 (two float4) into 4 packed-f16 dwords
__device__ __forceinline__ v4u pk8(const float* __restrict__ p) {
  v4f a = *(const v4f*)p;
  v4f b = *(const v4f*)(p + 4);
  v4u d = {pk2(a.x, a.y), pk2(a.z, a.w), pk2(b.x, b.y), pk2(b.z, b.w)};
  return d;
}

// fragment from two aligned 16B f16 chunks
__device__ __forceinline__ v16h frag_lds(const _Float16* __restrict__ row,
                                         int kc, int half8) {
  v4u lo = *(const v4u*)&row[kc + half8];
  v4u hi = *(const v4u*)&row[kc + 16 + half8];
  return __builtin_bit_cast(v16h,
         __builtin_shufflevector(lo, hi, 0, 1, 2, 3, 4, 5, 6, 7));
}

// fragment from 16 consecutive-pair f32 (aligned rows only)
__device__ __forceinline__ v16h frag_f32(const float* __restrict__ row,
                                         int kc, int half8) {
  v4u lo = pk8(&row[kc + half8]);
  v4u hi = pk8(&row[kc + 16 + half8]);
  return __builtin_bit_cast(v16h,
         __builtin_shufflevector(lo, hi, 0, 1, 2, 3, 4, 5, 6, 7));
}

// ---------------------------------------------------------------------------
// Implicit-GEMM convolution, one wave per 16(M) x 32(N) output tile (2 WMMAs).
// A: M = B*Ho*Wo rows (im2col), K = Ci*KF*KF, N = Co (multiple of 32).
// Weight element for (n, kk) is simply W[n*Ktot + kk] (OIHW is k-major).
// ---------------------------------------------------------------------------
template <int KF, int S, int P>
__global__ __launch_bounds__(32) void conv_wmma_kernel(
    const float* __restrict__ X, const float* __restrict__ W, float* __restrict__ Y,
    int Ci, int Hi, int Wi, int Co, int Ho, int Wo)
{
  constexpr int KK = KF * KF;
  const int lane = threadIdx.x;
  const int mt = blockIdx.x, nt = blockIdx.y;
  const int M = NB * Ho * Wo;
  const int Ktot = Ci * KK;
  const int half8 = (lane >> 4) << 3;

  const int m = mt * 16 + (lane & 15);
  int ib = 0, oh = 0, ow = 0;
  const bool mv = (m < M);
  if (mv) { ib = m / (Ho * Wo); int r2 = m % (Ho * Wo); oh = r2 / Wo; ow = r2 % Wo; }
  const int n0 = nt * 32 + (lane & 15);
  const float* __restrict__ w0 = W + (size_t)n0 * Ktot;
  const float* __restrict__ w1 = W + (size_t)(n0 + 16) * Ktot;

  // branchless im2col gather: unconditional load from clamped address + select
  auto aelem = [&](int kk) -> float {
    int ci = kk / KK;                    // constant divisor
    int rem = kk - ci * KK;
    int kr = rem / KF;
    int kw2 = rem - kr * KF;
    int ih = oh * S - P + kr, iw = ow * S - P + kw2;
    bool inb = mv & ((unsigned)ih < (unsigned)Hi) & ((unsigned)iw < (unsigned)Wi);
    int ihc = inb ? ih : 0, iwc = inb ? iw : 0;
    float vv = X[(((size_t)ib * Ci + ci) * Hi + ihc) * Wi + iwc];
    return inb ? vv : 0.f;
  };

  v8f acc0 = {0.f, 0.f, 0.f, 0.f, 0.f, 0.f, 0.f, 0.f};
  v8f acc1 = {0.f, 0.f, 0.f, 0.f, 0.f, 0.f, 0.f, 0.f};

#pragma unroll 1
  for (int kc = 0; kc < Ktot; kc += 32) {
    __builtin_prefetch(&w0[kc + 32], 0, 0);   // global_prefetch_b8, drop-safe
    __builtin_prefetch(&w1[kc + 32], 0, 0);
    v16h a, b0, b1;
    if constexpr (KF != 7) {
      // Ktot % 32 == 0 and weight rows 16B-aligned: vector weight fragments
      v8u au;
#pragma unroll
      for (int v = 0; v < 8; ++v) {
        int k0 = kc + kbase(v, half8);
        au[v] = pk2(aelem(k0), aelem(k0 + 1));
      }
      a  = __builtin_bit_cast(v16h, au);
      b0 = frag_f32(w0, kc, half8);
      b1 = frag_f32(w1, kc, half8);
    } else {
      // stem: Ktot = 147 (odd) -> scalar weight loads, full/tail chunk split
      v8u au, b0u, b1u;
      if (kc + 32 <= Ktot) {
#pragma unroll
        for (int v = 0; v < 8; ++v) {
          int k0 = kc + kbase(v, half8);
          au[v]  = pk2(aelem(k0), aelem(k0 + 1));
          b0u[v] = pk2(w0[k0], w0[k0 + 1]);
          b1u[v] = pk2(w1[k0], w1[k0 + 1]);
        }
      } else {
#pragma unroll
        for (int v = 0; v < 8; ++v) {
          int k0 = kc + kbase(v, half8);
          int kA = k0 < Ktot ? k0 : Ktot - 1;     // clamped valid addresses
          int kB = k0 + 1 < Ktot ? k0 + 1 : Ktot - 1;
          float a0 = aelem(kA), a1 = aelem(kB);
          float u0 = w0[kA], u1 = w0[kB];
          float t0 = w1[kA], t1 = w1[kB];
          bool q0 = (k0 < Ktot), q1 = (k0 + 1 < Ktot);
          au[v]  = pk2(q0 ? a0 : 0.f, q1 ? a1 : 0.f);
          b0u[v] = pk2(q0 ? u0 : 0.f, q1 ? u1 : 0.f);
          b1u[v] = pk2(q0 ? t0 : 0.f, q1 ? t1 : 0.f);
        }
      }
      a  = __builtin_bit_cast(v16h, au);
      b0 = __builtin_bit_cast(v16h, b0u);
      b1 = __builtin_bit_cast(v16h, b1u);
    }
    acc0 = __builtin_amdgcn_wmma_f32_16x16x32_f16(false, a, false, b0, (short)0, acc0,
                                                  false, false);
    acc1 = __builtin_amdgcn_wmma_f32_16x16x32_f16(false, a, false, b1, (short)0, acc1,
                                                  false, false);
  }

  // C/D layout: lane L, VGPR r -> (M = 8*(L/16)+r, N = L%16)
#pragma unroll
  for (int r = 0; r < 8; ++r) {
    int mm = mt * 16 + half8 + r;
    if (mm < M) {
      int bb = mm / (Ho * Wo); int r2 = mm % (Ho * Wo);
      int oh2 = r2 / Wo; int ow2 = r2 % Wo;
      size_t base = (((size_t)bb * Co) * Ho + oh2) * Wo + ow2;
      Y[base + (size_t)n0 * Ho * Wo] = acc0[r];
      Y[base + (size_t)(n0 + 16) * Ho * Wo] = acc1[r];
    }
  }
}

// ---------------------------------------------------------------------------
// Training-mode batch-norm: per-channel mean / inv-std over (B, H, W).
// ---------------------------------------------------------------------------
__global__ __launch_bounds__(256) void bn_stats_kernel(
    const float* __restrict__ X, float* __restrict__ mean, float* __restrict__ istd,
    int C, int HW)
{
  const int c = blockIdx.x;
  float s = 0.f, s2 = 0.f;
  for (int b = 0; b < NB; ++b) {
    const float* p = X + ((size_t)b * C + c) * HW;
    for (int i = threadIdx.x; i < HW; i += 256) {
      float v = p[i];
      s += v; s2 += v * v;
    }
  }
  __shared__ float sh[256], sh2[256];
  sh[threadIdx.x] = s; sh2[threadIdx.x] = s2;
  __syncthreads();
  for (int st = 128; st > 0; st >>= 1) {
    if (threadIdx.x < st) {
      sh[threadIdx.x]  += sh[threadIdx.x + st];
      sh2[threadIdx.x] += sh2[threadIdx.x + st];
    }
    __syncthreads();
  }
  if (threadIdx.x == 0) {
    float N = (float)(NB * HW);
    float m = sh[0] / N;
    float v = sh2[0] / N - m * m;
    mean[c] = m;
    istd[c] = rsqrtf(v + 1e-5f);
  }
}

// grid: (ceil(HW/256), NB*C). One block row per (b,c) plane: no per-element div.
__global__ __launch_bounds__(256) void bn_apply_kernel(
    const float* __restrict__ X, const float* __restrict__ Res, float* __restrict__ Y,
    const float* __restrict__ mean, const float* __restrict__ istd,
    const float* __restrict__ gam, const float* __restrict__ bet,
    int C, int HW, int relu)
{
  int p = blockIdx.x * 256 + threadIdx.x;
  if (p >= HW) return;
  int bc = blockIdx.y;
  int c = bc % C;
  size_t i = (size_t)bc * HW + p;
  float y = (X[i] - mean[c]) * istd[c] * gam[c] + bet[c];
  if (Res) y += Res[i];
  if (relu) y = fmaxf(y, 0.f);
  Y[i] = y;
}

__global__ __launch_bounds__(256) void maxpool_kernel(
    const float* __restrict__ X, float* __restrict__ Y,
    int C, int Hi, int Wi, int Ho, int Wo)
{
  int i = blockIdx.x * blockDim.x + threadIdx.x;
  int total = NB * C * Ho * Wo;
  if (i >= total) return;
  int ow = i % Wo; int t = i / Wo;
  int oh = t % Ho; t /= Ho;
  int c = t % C;  int b = t / C;
  float mx = -INFINITY;
  for (int dy = 0; dy < 3; ++dy)
    for (int dx = 0; dx < 3; ++dx) {
      int ih = oh * 2 - 1 + dy, iw = ow * 2 - 1 + dx;
      if (ih >= 0 && ih < Hi && iw >= 0 && iw < Wi)
        mx = fmaxf(mx, X[(((size_t)b * C + c) * Hi + ih) * Wi + iw]);
    }
  Y[i] = mx;
}

// avgpool -> h0 rows 0..3 (h buffer is [16][512], rows 4..15 stay zero)
__global__ __launch_bounds__(256) void avgpool_kernel(
    const float* __restrict__ X, float* __restrict__ h, int C, int HW)
{
  int i = blockIdx.x * blockDim.x + threadIdx.x;
  if (i >= NB * C) return;
  int b = i / C, c = i % C;
  float s = 0.f;
  for (int p = 0; p < HW; ++p) s += X[((size_t)b * C + c) * HW + p];
  h[b * HDIM + c] = s / (float)HW;
}

__global__ __launch_bounds__(256) void gru_init_kernel(
    float* __restrict__ h0, float* __restrict__ h1, unsigned* __restrict__ bar)
{
  int i = blockIdx.x * blockDim.x + threadIdx.x;
  if (i < 16 * HDIM) { h0[i] = 0.f; h1[i] = 0.f; }
  if (i == 0) *bar = 0u;
}

// Monotonic device-wide barrier (all 32 WGs resident).
__device__ __forceinline__ void grid_barrier(unsigned* bar, unsigned target) {
  __syncthreads();
  if (threadIdx.x == 0) {
    __threadfence();
    atomicAdd(bar, 1u);
    while (__hip_atomic_load(bar, __ATOMIC_ACQUIRE, __HIP_MEMORY_SCOPE_AGENT) < target)
      __builtin_amdgcn_s_sleep(1);
  }
  __syncthreads();
}

// ---------------------------------------------------------------------------
// Persistent GRU kernel. 32 WGs x 96 threads (3 waves). WG g owns h columns
// [g*16, g*16+16); wave w (= gate r/z/n) holds the matching 16x512 W_hh slice
// in LDS as f16. sH has 5 rows: rows 0..3 = h (staged per cell), row 4 = zeros,
// so the M-padded A fragment indexes hrow=min(mrow,4) with no per-element
// select; each fragment is two ds_load_b128. One cell = 16 LDS-resident
// WMMAs/wave + gate update + grid barrier. LDS: 48KB + 5KB + 0.75KB.
// ---------------------------------------------------------------------------
__global__ __launch_bounds__(96) void gru_kernel(
    const float* __restrict__ W_ih, const float* __restrict__ b_ih,
    const float* __restrict__ W_hh, const float* __restrict__ b_hh,
    const float* __restrict__ embed,
    float* __restrict__ h0, float* __restrict__ h1,
    float* __restrict__ gi_buf, float* __restrict__ out_buf,
    unsigned* __restrict__ bar)
{
  __shared__ __align__(16) _Float16 sW[48 * HDIM];  // [gate(3)][n(16)][k(512)]
  __shared__ __align__(16) _Float16 sH[5 * HDIM];   // rows 0..3 = h, row 4 = 0
  __shared__ float sGh[3][NB][16];

  const int tid  = threadIdx.x;
  const int w    = tid >> 5;             // gate index 0=r,1=z,2=n
  const int lane = tid & 31;
  const int g    = blockIdx.x;           // column-slice owner
  const int mrow = lane & 15;
  const int hrow = (mrow < NB) ? mrow : NB;   // zero row for padded M rows
  const int half8 = (lane >> 4) << 3;
  const float*    __restrict__ wir = W_ih + (size_t)(w * HDIM + g * 16 + mrow) * HDIM;
  const _Float16* __restrict__ hA  = &sH[hrow * HDIM];
  const _Float16* __restrict__ wB  = &sW[(w * 16 + mrow) * HDIM];

  // Preload W_hh slice -> LDS (two float4 in, one 16B LDS store out)
  for (int i = tid * 8; i < 48 * HDIM; i += 96 * 8) {
    int rn = i >> 9, k = i & (HDIM - 1);
    const float* src =
        W_hh + ((size_t)((rn >> 4) * HDIM + g * 16 + (rn & 15))) * HDIM + k;
    *(v4u*)&sW[i] = pk8(src);
  }
  // Stage x0 (= embed row 0, broadcast over batch rows) + zero row 4
  for (int i = tid * 8; i < 5 * HDIM; i += 96 * 8) {
    v4u val = {0u, 0u, 0u, 0u};
    if (i < NB * HDIM) val = pk8(embed + (i & (HDIM - 1)));
    *(v4u*)&sH[i] = val;
  }
  __syncthreads();

  // Prologue: gi[0] = x0 @ W_ih^T + b_ih ; out[0] = x0
  {
    v8f acc = {0.f, 0.f, 0.f, 0.f, 0.f, 0.f, 0.f, 0.f};
#pragma unroll 1
    for (int kc = 0; kc < HDIM; kc += 32) {
      v16h a  = frag_lds(hA, kc, half8);
      v16h bf = frag_f32(wir, kc, half8);
      acc = __builtin_amdgcn_wmma_f32_16x16x32_f16(false, a, false, bf, (short)0, acc,
                                                   false, false);
    }
    if (lane < 16) {
#pragma unroll
      for (int r = 0; r < NB; ++r)
        gi_buf[((size_t)0 * NB + r) * (3 * HDIM) + w * HDIM + g * 16 + lane] =
            acc[r] + b_ih[w * HDIM + g * 16 + lane];
    }
    if (tid < 64) {
      int bb = tid >> 4, j = tid & 15;
      out_buf[((size_t)0 * NB + bb) * HDIM + g * 16 + j] = embed[g * 16 + j];
    }
  }

  float* hcur = h0;                      // h0 holds feat (rows 0..3), rows 4..15 zero
  float* hnext = h1;
  unsigned epoch = 0;

#pragma unroll 1
  for (int t = 0; t < TMAX; ++t) {
#pragma unroll 1
    for (int k = 0; k <= t; ++k) {
      // stage h rows 0..3 into LDS f16; prior readers passed grid barrier
      __syncthreads();
      for (int i = tid * 8; i < NB * HDIM; i += 96 * 8)
        *(v4u*)&sH[i] = pk8(hcur + i);
      __syncthreads();

      // gh(gate w, cols g*16..+16) = h @ W_hh^T   (both operands LDS-resident)
      v8f acc = {0.f, 0.f, 0.f, 0.f, 0.f, 0.f, 0.f, 0.f};
#pragma unroll 1
      for (int kc = 0; kc < HDIM; kc += 32) {
        v16h a  = frag_lds(hA, kc, half8);
        v16h bf = frag_lds(wB, kc, half8);
        acc = __builtin_amdgcn_wmma_f32_16x16x32_f16(false, a, false, bf, (short)0, acc,
                                                     false, false);
      }
      if (lane < 16) {
#pragma unroll
        for (int r = 0; r < NB; ++r)
          sGh[w][r][lane] = acc[r] + b_hh[w * HDIM + g * 16 + lane];
      }
      __syncthreads();

      if (tid < 64) {
        int bb = tid >> 4, j = tid & 15, col = g * 16 + j;
        const float* gik = gi_buf + ((size_t)k * NB + bb) * (3 * HDIM);
        float rr = gik[col]            + sGh[0][bb][j];
        float zz = gik[HDIM + col]     + sGh[1][bb][j];
        float gn = gik[2 * HDIM + col];
        float r_ = 1.f / (1.f + __expf(-rr));
        float z_ = 1.f / (1.f + __expf(-zz));
        float n_ = tanhf(gn + r_ * sGh[2][bb][j]);
        float ho = hcur[bb * HDIM + col];
        hnext[bb * HDIM + col] = (1.f - z_) * n_ + z_ * ho;
      }

      ++epoch;
      grid_barrier(bar, epoch * GRU_WGS);
      float* tmp = hcur; hcur = hnext; hnext = tmp;
    }

    // gi[t+1] = h @ W_ih^T + b_ih ; out[t+1] = h   (all WG-local writes)
    {
      __syncthreads();
      for (int i = tid * 8; i < NB * HDIM; i += 96 * 8)
        *(v4u*)&sH[i] = pk8(hcur + i);
      __syncthreads();

      v8f acc = {0.f, 0.f, 0.f, 0.f, 0.f, 0.f, 0.f, 0.f};
#pragma unroll 1
      for (int kc = 0; kc < HDIM; kc += 32) {
        v16h a  = frag_lds(hA, kc, half8);
        v16h bf = frag_f32(wir, kc, half8);
        acc = __builtin_amdgcn_wmma_f32_16x16x32_f16(false, a, false, bf, (short)0, acc,
                                                     false, false);
      }
      if (lane < 16) {
#pragma unroll
        for (int r = 0; r < NB; ++r)
          gi_buf[((size_t)(t + 1) * NB + r) * (3 * HDIM) + w * HDIM + g * 16 + lane] =
              acc[r] + b_ih[w * HDIM + g * 16 + lane];
      }
      if (tid < 64) {
        int bb = tid >> 4, j = tid & 15;
        out_buf[((size_t)(t + 1) * NB + bb) * HDIM + g * 16 + j] =
            hcur[bb * HDIM + g * 16 + j];
      }
      __syncthreads();   // keep sH stable until all waves finished this GEMM
    }
  }
}

// ---------------------------------------------------------------------------
// logits[b,t,v] = out[t,b,:] . Wp[v,:] + bp[v] ; argmax over v (first-max ties)
// ---------------------------------------------------------------------------
__global__ __launch_bounds__(128) void proj_argmax_kernel(
    const float* __restrict__ ob, const float* __restrict__ Wp,
    const float* __restrict__ bp, float* __restrict__ out)
{
  const int tb = blockIdx.x;
  const int t = tb / NB, b = tb % NB;
  const float* h = ob + ((size_t)t * NB + b) * HDIM;
  __shared__ float sv[128];
  __shared__ int   si[128];
  float best = -INFINITY; int bidx = 0;
  for (int v = threadIdx.x; v < 80; v += 128) {
    float d = bp[v];
    for (int k = 0; k < HDIM; ++k) d += h[k] * Wp[(size_t)v * HDIM + k];
    if (d > best) { best = d; bidx = v; }
  }
  sv[threadIdx.x] = best; si[threadIdx.x] = bidx;
  __syncthreads();
  for (int st = 64; st > 0; st >>= 1) {
    if (threadIdx.x < st) {
      float v2 = sv[threadIdx.x + st]; int i2 = si[threadIdx.x + st];
      if (v2 > sv[threadIdx.x] || (v2 == sv[threadIdx.x] && i2 < si[threadIdx.x])) {
        sv[threadIdx.x] = v2; si[threadIdx.x] = i2;
      }
    }
    __syncthreads();
  }
  if (threadIdx.x == 0) out[b * (TMAX + 1) + t] = (float)si[0];
}

// ---------------------------------------------------------------------------
// Host orchestration
// ---------------------------------------------------------------------------
static inline void run_conv(const float* X, const float* W, float* Y,
                            int Ci, int Hi, int Wi, int Co, int Ho, int Wo,
                            int K, int S, hipStream_t s)
{
  int M = NB * Ho * Wo;
  dim3 grid((M + 15) / 16, Co / 32);
  if (K == 7)
    conv_wmma_kernel<7, 2, 3><<<grid, 32, 0, s>>>(X, W, Y, Ci, Hi, Wi, Co, Ho, Wo);
  else if (K == 3 && S == 1)
    conv_wmma_kernel<3, 1, 1><<<grid, 32, 0, s>>>(X, W, Y, Ci, Hi, Wi, Co, Ho, Wo);
  else if (K == 3 && S == 2)
    conv_wmma_kernel<3, 2, 1><<<grid, 32, 0, s>>>(X, W, Y, Ci, Hi, Wi, Co, Ho, Wo);
  else
    conv_wmma_kernel<1, 2, 0><<<grid, 32, 0, s>>>(X, W, Y, Ci, Hi, Wi, Co, Ho, Wo);
}

static inline void run_bnstats(const float* X, float* mean, float* istd,
                               int C, int HW, hipStream_t s)
{
  bn_stats_kernel<<<C, 256, 0, s>>>(X, mean, istd, C, HW);
}

static inline void run_bnapply(const float* X, const float* Res, float* Y,
                               const float* mean, const float* istd,
                               const float* g, const float* b,
                               int C, int HW, int relu, hipStream_t s)
{
  dim3 grid((HW + 255) / 256, NB * C);
  bn_apply_kernel<<<grid, 256, 0, s>>>(X, Res, Y, mean, istd, g, b, C, HW, relu);
}

extern "C" void kernel_launch(void* const* d_in, const int* in_sizes, int n_in,
                              void* d_out, int out_size, void* d_ws, size_t ws_size,
                              hipStream_t stream)
{
  (void)in_sizes; (void)n_in; (void)out_size;
  // Input order = setup_inputs() dict insertion order, depth-first:
  // img, stem{w,g,b}, stages[4][2]{c1,g1,b1,c2,g2,b2[,cs,gs,bs]},
  // gru{W_ih,b_ih,W_hh,b_hh}, proj{W,b}, embed
  int pi = 0;
  auto nxt = [&]() { return (const float*)d_in[pi++]; };
  const float* img = nxt();
  const float* stem_w = nxt(); const float* stem_g = nxt(); const float* stem_b = nxt();

  const float *c1[8], *g1[8], *b1[8], *c2[8], *g2[8], *b2[8];
  const float *wcs[8], *gcs[8], *bcs[8];
  bool hascs[8];
  {
    int k = 0;
    for (int si = 0; si < 4; ++si)
      for (int bi = 0; bi < 2; ++bi, ++k) {
        c1[k] = nxt(); g1[k] = nxt(); b1[k] = nxt();
        c2[k] = nxt(); g2[k] = nxt(); b2[k] = nxt();
        hascs[k] = (bi == 0 && si > 0);
        if (hascs[k]) { wcs[k] = nxt(); gcs[k] = nxt(); bcs[k] = nxt(); }
        else          { wcs[k] = gcs[k] = bcs[k] = nullptr; }
      }
  }
  const float* W_ih = nxt(); const float* bih = nxt();
  const float* W_hh = nxt(); const float* bhh = nxt();
  const float* projW = nxt(); const float* projB = nxt();
  const float* embed = nxt();

  // Workspace carving
  char* wsp = (char*)d_ws;
  auto carve = [&](size_t bytes) -> void* {
    void* p = (void*)wsp;
    wsp += (bytes + 255) & ~(size_t)255;
    return p;
  };
  const size_t ACT = (size_t)NB * 64 * 112 * 112;   // largest activation (floats)
  float* bA = (float*)carve(ACT * 4);
  float* bB = (float*)carve(ACT * 4);
  float* bC = (float*)carve(ACT * 4);
  float* meanA = (float*)carve(512 * 4); float* istdA = (float*)carve(512 * 4);
  float* meanB = (float*)carve(512 * 4); float* istdB = (float*)carve(512 * 4);
  float* h0 = (float*)carve(16 * HDIM * 4);
  float* h1 = (float*)carve(16 * HDIM * 4);
  float* gi = (float*)carve((size_t)(TMAX + 1) * NB * 3 * HDIM * 4);
  float* ob = (float*)carve((size_t)(TMAX + 1) * NB * HDIM * 4);
  unsigned* bar = (unsigned*)carve(256);
  if ((size_t)(wsp - (char*)d_ws) > ws_size) return;   // not enough scratch

  // Reset persistent-kernel state each call (graph-replay safe)
  gru_init_kernel<<<32, 256, 0, stream>>>(h0, h1, bar);

  // Stem: conv7x7/2 -> BN(train) -> relu -> maxpool3x3/2
  run_conv(img, stem_w, bA, 3, 224, 224, 64, 112, 112, 7, 2, stream);
  run_bnstats(bA, meanA, istdA, 64, 112 * 112, stream);
  run_bnapply(bA, nullptr, bA, meanA, istdA, stem_g, stem_b, 64, 112 * 112, 1, stream);
  {
    int total = NB * 64 * 56 * 56;
    maxpool_kernel<<<(total + 255) / 256, 256, 0, stream>>>(bA, bB, 64, 112, 112, 56, 56);
  }

  // Residual stages
  float *P = bB, *Q = bA, *R = bC;
  int hin = 56, ic = 64;
  const int chans[4] = {64, 128, 256, 512};
  int k = 0;
  for (int si = 0; si < 4; ++si) {
    int c = chans[si];
    for (int bi = 0; bi < 2; ++bi, ++k) {
      int s = (bi == 0 && si > 0) ? 2 : 1;
      int icb = (bi == 0) ? ic : c;
      int ho = (s == 1) ? hin : ((hin - 1) / 2 + 1);
      int hw = ho * ho;
      // main path
      run_conv(P, c1[k], Q, icb, hin, hin, c, ho, ho, 3, s, stream);
      run_bnstats(Q, meanB, istdB, c, hw, stream);
      run_bnapply(Q, nullptr, Q, meanB, istdB, g1[k], b1[k], c, hw, 1, stream);
      run_conv(Q, c2[k], R, c, ho, ho, c, ho, ho, 3, 1, stream);
      run_bnstats(R, meanA, istdA, c, hw, stream);
      if (hascs[k]) {
        // projected shortcut (1x1/s) -> BN, then fused add+relu into P
        run_conv(P, wcs[k], Q, icb, hin, hin, c, ho, ho, 1, 2, stream);
        run_bnstats(Q, meanB, istdB, c, hw, stream);
        run_bnapply(Q, nullptr, Q, meanB, istdB, gcs[k], bcs[k], c, hw, 0, stream);
        run_bnapply(R, Q, P, meanA, istdA, g2[k], b2[k], c, hw, 1, stream);
        // new x stays in P
      } else {
        run_bnapply(R, P, Q, meanA, istdA, g2[k], b2[k], c, hw, 1, stream);
        float* t = P; P = Q; Q = t;       // new x in former Q
      }
      hin = ho;
    }
    ic = c;
  }

  // Global average pool -> feat into h0 rows 0..3
  avgpool_kernel<<<(NB * HDIM + 255) / 256, 256, 0, stream>>>(P, h0, HDIM, 7 * 7);

  // Triangular GRU decode (persistent, 32 WGs x 3 waves)
  gru_kernel<<<GRU_WGS, 96, 0, stream>>>(W_ih, bih, W_hh, bhh, embed,
                                         h0, h1, gi, ob, bar);

  // Projection + argmax -> d_out [B, T+1] (indices as float)
  proj_argmax_kernel<<<(TMAX + 1) * NB, 128, 0, stream>>>(ob, projW, projB,
                                                          (float*)d_out);
}
